// GCMC_82403242541302
// MI455X (gfx1250) — matrix operand order repaired and checked by reading, not compiled, not run on hardware
//
#include <hip/hip_runtime.h>

// ---------------- problem constants (match reference) ----------------
constexpr int cNU = 100000, cNI = 20000, cFU = 2000, cFI = 2000;
constexpr int cD = 320, cR = 5, cMSG = 64, cE = 300000, cED = 500000;
constexpr int cH1 = 200, cH2 = 40, cH3 = 640; // MLP pyramid 2000->200->40->640->320

typedef __attribute__((ext_vector_type(16))) __bf16 v16bf;
typedef __attribute__((ext_vector_type(8)))  float  v8f;

enum { RELU_OUT = 1, RELU_IN = 2 };

// LDS tile geometry (padded strides -> conflict-free 64-bank access)
#define A_STRIDE 36   // floats per A row in LDS (32 used + 4 pad)
#define B_STRIDE 40   // bf16 per Bt row in LDS (32 used + 8 pad)

// ---------------------------------------------------------------------
// C[M,N] = act( act_in(A[M,K]) @ W[K,N] + bias[N] )
// Block = 128 threads (4 waves); block tile 64x32; wave tile 16x32
// (two v_wmma_f32_16x16x32_bf16 sharing one A fragment per K-step).
// Double-buffered LDS pipeline: iteration t issues
// GLOBAL_LOAD_ASYNC_TO_LDS_B128 (ASYNCcnt) for tile t+1 while computing
// tile t, then a single s_wait_asynccnt + barrier per iteration.
// ---------------------------------------------------------------------
__global__ __launch_bounds__(128)
void gemm_wmma_bf16(const float* __restrict__ A, const float* __restrict__ W,
                    const float* __restrict__ bias, float* __restrict__ C,
                    int Mr, int Kd, int Nc, int flags)
{
    __shared__ float  Af[2][64][A_STRIDE];   // fp32 A tiles (2 x 9216 B)
    __shared__ __bf16 Bt[2][32][B_STRIDE];   // bf16 B tiles, transposed (2 x 2560 B)

    const int tid  = threadIdx.x;
    const int lane = tid & 31;
    const int wave = tid >> 5;
    const int mblk = blockIdx.y * 64;            // block row base
    const int n0   = blockIdx.x * 32;            // block col base
    const int m0   = mblk + wave * 16;           // this wave's row base

    const int  mr  = lane & 15;
    const bool hi  = (lane >= 16);
    const bool rin = (flags & RELU_IN) != 0;

    // --- staging roles ---
    const int ar  = tid >> 1;                // A: row within block tile (0..63)
    const int ah  = (tid & 1) * 16;          // A: 16-float half of the 32-col row
    const int agr = mblk + ar;               // A: global row
    const bool arow_ok = (agr < Mr);

    const int bkk = tid >> 2;                // B: k within K-step (0..31)
    const int bnn = (tid & 3) * 8;           // B: 8-col group

    auto stageA = [&](int k0, int buf) {
        const float* gp = A + (size_t)agr * Kd + k0 + ah;
        if ((k0 + 32 <= Kd) && arow_ok) {
            unsigned lofs = (unsigned)(uintptr_t)&Af[buf][ar][ah];
            unsigned long long ga = (unsigned long long)(uintptr_t)gp;
            asm volatile("global_load_async_to_lds_b128 %0, %1, off"
                         :: "v"(lofs),      "v"(ga)      : "memory");
            asm volatile("global_load_async_to_lds_b128 %0, %1, off"
                         :: "v"(lofs + 16), "v"(ga + 16) : "memory");
            asm volatile("global_load_async_to_lds_b128 %0, %1, off"
                         :: "v"(lofs + 32), "v"(ga + 32) : "memory");
            asm volatile("global_load_async_to_lds_b128 %0, %1, off"
                         :: "v"(lofs + 48), "v"(ga + 48) : "memory");
            __builtin_prefetch(gp + 32, 0, 1);   // K-block after next -> global_prefetch_b8
        } else {
#pragma unroll
            for (int j = 0; j < 16; ++j) {
                int k = k0 + ah + j;
                Af[buf][ar][ah + j] = (arow_ok && k < Kd) ? gp[j] : 0.f;
            }
        }
    };

    auto stageB = [&](int k0, int buf) {
        const int gk = k0 + bkk;
        const float* wp = W + (size_t)gk * Nc + n0 + bnn;
        const bool kok = (gk < Kd);
#pragma unroll
        for (int c = 0; c < 8; ++c) {
            float v = (kok && (n0 + bnn + c) < Nc) ? wp[c] : 0.f;
            Bt[buf][bnn + c][bkk] = (__bf16)v;
        }
    };

    v8f acc0 = {0.f, 0.f, 0.f, 0.f, 0.f, 0.f, 0.f, 0.f};
    v8f acc1 = {0.f, 0.f, 0.f, 0.f, 0.f, 0.f, 0.f, 0.f};

    // prologue: stage first tile into buffer 0
    stageA(0, 0);
    stageB(0, 0);
    asm volatile("s_wait_asynccnt 0x0" ::: "memory");
    __syncthreads();

    int parity = 0;
    for (int k0 = 0; k0 < Kd; k0 += 32) {
        const int nk = k0 + 32;
        if (nk < Kd) {                       // stage next tile into the other buffer
            stageA(nk, parity ^ 1);
            stageB(nk, parity ^ 1);
        }

        // ---------- A fragment: lanes0-15 K 0-7 & 16-23; lanes16-31 K 8-15 & 24-31
        const int kb0 = hi ? 8 : 0;
        const int kb1 = hi ? 24 : 16;
        const float* ap = &Af[parity][wave * 16 + mr][0];
        float4 f0 = *(const float4*)(ap + kb0);
        float4 f1 = *(const float4*)(ap + kb0 + 4);
        float4 f2 = *(const float4*)(ap + kb1);
        float4 f3 = *(const float4*)(ap + kb1 + 4);
        if (rin) {
            f0.x = fmaxf(f0.x, 0.f); f0.y = fmaxf(f0.y, 0.f); f0.z = fmaxf(f0.z, 0.f); f0.w = fmaxf(f0.w, 0.f);
            f1.x = fmaxf(f1.x, 0.f); f1.y = fmaxf(f1.y, 0.f); f1.z = fmaxf(f1.z, 0.f); f1.w = fmaxf(f1.w, 0.f);
            f2.x = fmaxf(f2.x, 0.f); f2.y = fmaxf(f2.y, 0.f); f2.z = fmaxf(f2.z, 0.f); f2.w = fmaxf(f2.w, 0.f);
            f3.x = fmaxf(f3.x, 0.f); f3.y = fmaxf(f3.y, 0.f); f3.z = fmaxf(f3.z, 0.f); f3.w = fmaxf(f3.w, 0.f);
        }
        v16bf a;
        a[0]  = (__bf16)f0.x; a[1]  = (__bf16)f0.y; a[2]  = (__bf16)f0.z; a[3]  = (__bf16)f0.w;
        a[4]  = (__bf16)f1.x; a[5]  = (__bf16)f1.y; a[6]  = (__bf16)f1.z; a[7]  = (__bf16)f1.w;
        a[8]  = (__bf16)f2.x; a[9]  = (__bf16)f2.y; a[10] = (__bf16)f2.z; a[11] = (__bf16)f2.w;
        a[12] = (__bf16)f3.x; a[13] = (__bf16)f3.y; a[14] = (__bf16)f3.z; a[15] = (__bf16)f3.w;

        // ---------- B fragments (two 16-col groups share the A fragment)
        const int kb = hi ? 8 : 0;
        union { uint4 q[2]; v16bf v; } ub0, ub1;
        ub0.q[0] = *(const uint4*)&Bt[parity][mr][kb];
        ub0.q[1] = *(const uint4*)&Bt[parity][mr][kb + 16];
        ub1.q[0] = *(const uint4*)&Bt[parity][16 + mr][kb];
        ub1.q[1] = *(const uint4*)&Bt[parity][16 + mr][kb + 16];

        acc0 = __builtin_amdgcn_wmma_f32_16x16x32_bf16(
            false, a, false, ub0.v, (short)0, acc0, false, false);
        acc1 = __builtin_amdgcn_wmma_f32_16x16x32_bf16(
            false, a, false, ub1.v, (short)0, acc1, false, false);

        asm volatile("s_wait_asynccnt 0x0" ::: "memory");
        __syncthreads();
        parity ^= 1;
    }

    // ---- epilogue: VGPR r holds M = r + (lane>=16 ? 8 : 0), N = lane&15
    const int mb   = m0 + (hi ? 8 : 0);
    const int col0 = n0 + mr;
    const int col1 = n0 + 16 + mr;
    const float b0 = (bias != nullptr && col0 < Nc) ? bias[col0] : 0.f;
    const float b1 = (bias != nullptr && col1 < Nc) ? bias[col1] : 0.f;
#pragma unroll
    for (int r = 0; r < 8; ++r) {
        int m = mb + r;
        if (m < Mr) {
            if (col0 < Nc) {
                float v = acc0[r] + b0;
                if (flags & RELU_OUT) v = fmaxf(v, 0.f);
                C[(size_t)m * Nc + col0] = v;
            }
            if (col1 < Nc) {
                float v = acc1[r] + b1;
                if (flags & RELU_OUT) v = fmaxf(v, 0.f);
                C[(size_t)m * Nc + col1] = v;
            }
        }
    }
}

// ---------------------------------------------------------------------
__global__ void zero_f32(float* __restrict__ p, long long n)
{
    long long t = (long long)blockIdx.x * blockDim.x + threadIdx.x;
    if (t < n) p[t] = 0.f;
}

__global__ void count_deg(const int* __restrict__ su, const int* __restrict__ di,
                          float* __restrict__ deg_u, float* __restrict__ deg_i, int E)
{
    int e = blockIdx.x * blockDim.x + threadIdx.x;
    if (e >= E) return;
    atomicAdd(&deg_u[su[e]], 1.f);
    atomicAdd(&deg_i[di[e]], 1.f);
}

// per-rating message scatter with symmetric degree norm, written directly
// into the stack-concat layout h[node, r*64 + m]
__global__ void scatter_msg(const float* __restrict__ msg_u, const float* __restrict__ msg_i,
                            const int* __restrict__ su, const int* __restrict__ di,
                            const float* __restrict__ deg_u, const float* __restrict__ deg_i,
                            float* __restrict__ hu, float* __restrict__ hi,
                            int r, int E)
{
    long long t = (long long)blockIdx.x * blockDim.x + threadIdx.x;
    long long total = (long long)E * cMSG;
    if (t >= total) return;
    int e = (int)(t >> 6);
    int m = (int)(t & 63);
    int u = su[e], i = di[e];
    float en = rsqrtf(fmaxf(deg_u[u], 1.f)) * rsqrtf(fmaxf(deg_i[i], 1.f));
    atomicAdd(&hi[(size_t)i * cD + r * cMSG + m], msg_u[(size_t)u * cMSG + m] * en);
    atomicAdd(&hu[(size_t)u * cD + r * cMSG + m], msg_i[(size_t)i * cMSG + m] * en);
}

// BiDecoder refactored: basis[e,k] = dot(T_k[dec_u[e]], i_emb[dec_i[e]]);
// T_k = u_emb @ P[k] precomputed by GEMM. One wave32 per edge.
__global__ __launch_bounds__(256)
void decoder_k(const float* __restrict__ T0, const float* __restrict__ T1,
               const float* __restrict__ iF,
               const int* __restrict__ du, const int* __restrict__ dj,
               const float* __restrict__ combW, const float* __restrict__ combB,
               float* __restrict__ out, int Ed)
{
    int lane = threadIdx.x & 31;
    int e = blockIdx.x * (blockDim.x >> 5) + (threadIdx.x >> 5);
    if (e >= Ed) return;
    int u = du[e], it = dj[e];
    const float* t0 = T0 + (size_t)u  * cD;
    const float* t1 = T1 + (size_t)u  * cD;
    const float* ie = iF + (size_t)it * cD;
    float s0 = 0.f, s1 = 0.f;
#pragma unroll
    for (int d = 0; d < cD; d += 32) {
        float iv = ie[d + lane];
        s0 += t0[d + lane] * iv;
        s1 += t1[d + lane] * iv;
    }
#pragma unroll
    for (int off = 16; off > 0; off >>= 1) {
        s0 += __shfl_xor(s0, off, 32);
        s1 += __shfl_xor(s1, off, 32);
    }
    if (lane < cR)
        out[(size_t)e * cR + lane] = s0 * combW[lane] + s1 * combW[cR + lane] + combB[lane];
}

// ---------------------------------------------------------------------
static inline void launch_gemm(const float* A, const float* W, const float* b, float* C,
                               int Mr, int Kd, int Nc, int flags, hipStream_t s)
{
    dim3 grid((Nc + 31) / 32, (Mr + 63) / 64);
    gemm_wmma_bf16<<<grid, 128, 0, s>>>(A, W, b, C, Mr, Kd, Nc, flags);
}

static inline void launch_zero(float* p, long long n, hipStream_t s)
{
    zero_f32<<<(unsigned)((n + 255) / 256), 256, 0, s>>>(p, n);
}

extern "C" void kernel_launch(void* const* d_in, const int* in_sizes, int n_in,
                              void* d_out, int out_size, void* d_ws, size_t ws_size,
                              hipStream_t stream)
{
    (void)in_sizes; (void)n_in; (void)out_size; (void)ws_size;

    // ---- inputs, in setup_inputs() dict order (lists flattened) ----
    const float* u_feat = (const float*)d_in[0];
    const float* i_feat = (const float*)d_in[1];
    const float* uW[4]  = {(const float*)d_in[2], (const float*)d_in[3],
                           (const float*)d_in[4], (const float*)d_in[5]};
    const float* ub[4]  = {(const float*)d_in[6], (const float*)d_in[7],
                           (const float*)d_in[8], (const float*)d_in[9]};
    const float* iW[4]  = {(const float*)d_in[10], (const float*)d_in[11],
                           (const float*)d_in[12], (const float*)d_in[13]};
    const float* ib[4]  = {(const float*)d_in[14], (const float*)d_in[15],
                           (const float*)d_in[16], (const float*)d_in[17]};
    const float* Wencu  = (const float*)d_in[18];   // [R,D,M]
    const float* Wenci  = (const float*)d_in[19];   // [R,D,M]
    const float* ufcW   = (const float*)d_in[20];
    const float* ufcB   = (const float*)d_in[21];
    const float* ifcW   = (const float*)d_in[22];
    const float* ifcB   = (const float*)d_in[23];
    const float* luW    = (const float*)d_in[24];
    const float* luB    = (const float*)d_in[25];
    const float* liW    = (const float*)d_in[26];
    const float* liB    = (const float*)d_in[27];
    const float* Pmat   = (const float*)d_in[28];   // [K,D,D]
    const float* combW  = (const float*)d_in[29];   // [K,R]
    const float* combB  = (const float*)d_in[30];   // [R]
    const int*   enc_u  = (const int*)d_in[31];     // [R,E]
    const int*   enc_i  = (const int*)d_in[32];     // [R,E]
    const int*   dec_u  = (const int*)d_in[33];     // [ED]
    const int*   dec_i  = (const int*)d_in[34];     // [ED]
    float*       out    = (float*)d_out;            // [ED,R]

    // ---- workspace carve-out (floats); h1/h3 share a region (disjoint lifetimes)
    float* ws = (float*)d_ws;
    size_t o = 0;
    auto alloc = [&](size_t nf) { size_t c = o; o += nf; return c; };
    const size_t u_h13 = alloc((size_t)cNU * cH3);
    const size_t u_h2  = alloc((size_t)cNU * cH2);
    const size_t uf_   = alloc((size_t)cNU * cD);
    const size_t i_h13 = alloc((size_t)cNI * cH3);
    const size_t i_h2  = alloc((size_t)cNI * cH2);
    const size_t itf_  = alloc((size_t)cNI * cD);
    const size_t degU  = alloc((size_t)cNU);
    const size_t degI  = alloc((size_t)cNI);
    const size_t msgU  = alloc((size_t)cNU * cMSG);
    const size_t msgI  = alloc((size_t)cNI * cMSG);
    const size_t huSt  = alloc((size_t)cNU * cD);
    const size_t hiSt  = alloc((size_t)cNI * cD);
    const size_t uA    = alloc((size_t)cNU * cD);
    const size_t iA    = alloc((size_t)cNI * cD);
    const size_t uF    = alloc((size_t)cNU * cD);
    const size_t iF    = alloc((size_t)cNI * cD);
    const size_t T0    = alloc((size_t)cNU * cD);
    const size_t T1    = alloc((size_t)cNU * cD);

    // ---- 1) feature MLPs ----
    launch_gemm(u_feat,     uW[0], ub[0], ws + u_h13, cNU, cFU, cH1, RELU_OUT, stream);
    launch_gemm(ws + u_h13, uW[1], ub[1], ws + u_h2,  cNU, cH1, cH2, RELU_OUT, stream);
    launch_gemm(ws + u_h2,  uW[2], ub[2], ws + u_h13, cNU, cH2, cH3, RELU_OUT, stream);
    launch_gemm(ws + u_h13, uW[3], ub[3], ws + uf_,   cNU, cH3, cD,  RELU_OUT, stream);

    launch_gemm(i_feat,     iW[0], ib[0], ws + i_h13, cNI, cFI, cH1, RELU_OUT, stream);
    launch_gemm(ws + i_h13, iW[1], ib[1], ws + i_h2,  cNI, cH1, cH2, RELU_OUT, stream);
    launch_gemm(ws + i_h2,  iW[2], ib[2], ws + i_h13, cNI, cH2, cH3, RELU_OUT, stream);
    launch_gemm(ws + i_h13, iW[3], ib[3], ws + itf_,  cNI, cH3, cD,  RELU_OUT, stream);

    // ---- 2) per-rating GCMC message passing into stack-concat layout ----
    launch_zero(ws + huSt, (long long)cNU * cD, stream);
    launch_zero(ws + hiSt, (long long)cNI * cD, stream);
    for (int r = 0; r < cR; ++r) {
        launch_zero(ws + degU, cNU, stream);
        launch_zero(ws + degI, cNI, stream);
        count_deg<<<(cE + 255) / 256, 256, 0, stream>>>(
            enc_u + (size_t)r * cE, enc_i + (size_t)r * cE, ws + degU, ws + degI, cE);
        launch_gemm(ws + uf_,  Wencu + (size_t)r * cD * cMSG, nullptr, ws + msgU,
                    cNU, cD, cMSG, 0, stream);
        launch_gemm(ws + itf_, Wenci + (size_t)r * cD * cMSG, nullptr, ws + msgI,
                    cNI, cD, cMSG, 0, stream);
        long long tot = (long long)cE * cMSG;
        scatter_msg<<<(unsigned)((tot + 255) / 256), 256, 0, stream>>>(
            ws + msgU, ws + msgI, enc_u + (size_t)r * cE, enc_i + (size_t)r * cE,
            ws + degU, ws + degI, ws + huSt, ws + hiSt, r, cE);
    }

    // ---- 3) post-aggregation dense layers ----
    launch_gemm(ws + huSt, ufcW, ufcB, ws + uA, cNU, cD, cD, RELU_IN | RELU_OUT, stream);
    launch_gemm(ws + hiSt, ifcW, ifcB, ws + iA, cNI, cD, cD, RELU_IN | RELU_OUT, stream);
    launch_gemm(ws + uA,   luW,  luB,  ws + uF, cNU, cD, cD, 0, stream);
    launch_gemm(ws + iA,   liW,  liB,  ws + iF, cNI, cD, cD, 0, stream);

    // ---- 4) decoder: per-node bilinear transforms (GEMM) + per-edge dots ----
    launch_gemm(ws + uF, Pmat,                   nullptr, ws + T0, cNU, cD, cD, 0, stream);
    launch_gemm(ws + uF, Pmat + (size_t)cD * cD, nullptr, ws + T1, cNU, cD, cD, 0, stream);
    decoder_k<<<(cED + 7) / 8, 256, 0, stream>>>(
        ws + T0, ws + T1, ws + iF, dec_u, dec_i, combW, combB, out, cED);
}
